// LSTMModel_19920058319316
// MI455X (gfx1250) — compile-verified, matrix operand order checked by reading, not run on hardware
//
#include <hip/hip_runtime.h>

typedef __bf16 bf16_t;
typedef __attribute__((ext_vector_type(16))) __bf16 v16bf;
typedef __attribute__((ext_vector_type(8)))  float  v8f;

union Frag16 { unsigned u[8]; v16bf v; };

#define LDS_AS __attribute__((address_space(3)))

// Problem dims
constexpr int B_  = 4;
constexpr int S_  = 2048;
constexpr int D_  = 1024;
constexpr int H_  = 1024;
constexpr int V_  = 4096;
constexpr int G4H = 4096;   // 4*H
constexpr int HID = 1024;

// Recurrence config: 32 persistent WGs, each owns 32 hidden units (128 W_hh rows) in LDS
constexpr int REC_WGS   = 32;
constexpr int UNITS     = H_ / REC_WGS;     // 32
constexpr int REC_ROWS  = 4 * UNITS;        // 128
constexpr int LDK_REC   = 1024 + 8;         // padded row stride (bf16 elems); 2064B = 129*16 OK
constexpr int REC_SMEM  = REC_ROWS * LDK_REC * 2   // W_hh slice (bf16)
                        + 16 * LDK_REC * 2          // h padded to 16 rows (bf16)
                        + REC_ROWS * 16 * 4         // gate fragments (f32)
                        + REC_ROWS * 4;             // cell state (f32)  => 305,920 B < 320 KB

// GEMM tile config: WG = 8 waves, wave tile 32x64, WG tile 256x64, double-buffered LDS
constexpr int GA_ROWS = 256;
constexpr int GB_ROWS = 64;
constexpr int GLDK    = 40;                       // 32 K + 8 pad (80 B row = 5*16 aligned)
constexpr int GA_ELEM = GA_ROWS * GLDK;           // 10240 bf16
constexpr int GB_ELEM = GB_ROWS * GLDK;           // 2560  bf16
constexpr int GBUF    = GA_ELEM + GB_ELEM;        // per-buffer elems (25600 B)

// Workspace layout (bytes)
constexpr size_t OFF_LANGBF = 0;                                   // 4*2048*1024 bf16
constexpr size_t OFF_WIHBF  = OFF_LANGBF + (size_t)B_*S_*D_*2;     // 4096*1024 bf16
constexpr size_t OFF_WHHBF  = OFF_WIHBF  + (size_t)G4H*D_*2;
constexpr size_t OFF_XW     = OFF_WHHBF  + (size_t)G4H*H_*2;       // 8192*4096 f32
constexpr size_t OFF_HBUF   = OFF_XW     + (size_t)S_*B_*G4H*4;    // 2*4*1024 bf16 ping-pong
constexpr size_t OFF_HF32   = OFF_HBUF   + (size_t)2*B_*H_*2;
constexpr size_t OFF_HID    = OFF_HF32   + (size_t)B_*H_*4;
constexpr size_t OFF_BAR    = OFF_HID    + (size_t)B_*HID*4;

// ---- CDNA5 async copy helpers (ASYNCcnt-tracked, no VGPR round trip) ----
__device__ __forceinline__ unsigned lds_off(const void* p) {
  return (unsigned)(size_t)(LDS_AS const char*)p;   // addrspacecast generic->LDS strips aperture
}
__device__ __forceinline__ void async_b128(const void* gsrc, const void* ldst) {
  unsigned l = lds_off(ldst);
  unsigned long long g = (unsigned long long)(size_t)gsrc;
  asm volatile("global_load_async_to_lds_b128 %0, %1, off" :: "v"(l), "v"(g) : "memory");
}
__device__ __forceinline__ void wait_async0() {
  asm volatile("s_wait_asynccnt 0x0" ::: "memory");
}

// ---- fragment load: 16-bit A/B layout, K-pairs are contiguous 32-bit LDS loads ----
__device__ __forceinline__ v16bf load_frag(const bf16_t* lds, int rowBase, int kBase, int stride) {
  const int lane = threadIdx.x & 31;
  const int row  = rowBase + (lane & 15);
  const int half = lane >> 4;
  Frag16 f;
#pragma unroll
  for (int i = 0; i < 8; ++i) {
    int k = kBase + ((i >> 2) << 4) + (half << 3) + ((i & 3) << 1);
    f.u[i] = *(const unsigned*)(lds + row * stride + k);
  }
  return f.v;
}

// ---- fp32 -> bf16 conversion ----
__global__ void cvt_bf16_kernel(const float* __restrict__ src, bf16_t* __restrict__ dst, int n) {
  int i = blockIdx.x * blockDim.x + threadIdx.x;
  int stride = gridDim.x * blockDim.x;
  for (; i < n; i += stride) dst[i] = (bf16_t)src[i];
}

// ---- init: zero grid barrier, seed h ping-pong buffer 0 with h0 ----
__global__ void init_kernel(const float* __restrict__ h0, bf16_t* __restrict__ hbuf,
                            unsigned* __restrict__ bar) {
  int i = blockIdx.x * blockDim.x + threadIdx.x;
  if (i == 0) { bar[0] = 0u; bar[1] = 0u; }
  if (i < B_ * H_) hbuf[i] = (bf16_t)h0[i];
}

// ---- Kernel 2: xw[s*B+b][g] = x[s][b][:] @ W_ih[g][:] + b_ih[g]  (bf16 WMMA, fp32 acc) ----
// Double-buffered LDS staging via global_load_async_to_lds_b128.
__global__ __launch_bounds__(256) void gemm_xw_kernel(
    const bf16_t* __restrict__ langbf,   // [B][S][D] bf16
    const bf16_t* __restrict__ wihbf,    // [4H][D] bf16
    const float* __restrict__ b_ih,
    float* __restrict__ xw)
{
  __shared__ bf16_t SM[2 * GBUF];
  const int mBase = blockIdx.x * GA_ROWS;
  const int nBase = blockIdx.y * GB_ROWS;
  const int tid   = threadIdx.x;
  const int wave  = tid >> 5;
  const int lane  = tid & 31;

  v8f acc[2][4];
#pragma unroll
  for (int mt = 0; mt < 2; ++mt)
#pragma unroll
    for (int nt = 0; nt < 4; ++nt) acc[mt][nt] = (v8f){0,0,0,0,0,0,0,0};

  // stage K-chunk kc into buffer `buf` (each lane moves 16B chunks: 8 bf16 along K)
  auto stage = [&](int buf, int kc) {
    bf16_t* Asl = SM + buf * GBUF;
    bf16_t* Bsl = Asl + GA_ELEM;
#pragma unroll
    for (int i = 0; i < 4; ++i) {            // A: 256 rows x 4 chunks
      int c = tid + i * 256;
      int row = c >> 2, cpos = c & 3;
      int m = mBase + row, s = m >> 2, b = m & 3;
      async_b128(langbf + ((size_t)b * S_ + s) * D_ + kc + cpos * 8,
                 Asl + row * GLDK + cpos * 8);
    }
    {                                        // B: 64 rows x 4 chunks
      int row = tid >> 2, cpos = tid & 3;
      async_b128(wihbf + (size_t)(nBase + row) * D_ + kc + cpos * 8,
                 Bsl + row * GLDK + cpos * 8);
    }
  };

  int buf = 0;
  stage(buf, 0);
  wait_async0();
  __syncthreads();

  for (int kc = 0; kc < D_; kc += 32) {
    if (kc + 32 < D_) stage(buf ^ 1, kc + 32);   // overlap next-chunk DMA with WMMAs

    const bf16_t* Asl = SM + buf * GBUF;
    const bf16_t* Bsl = Asl + GA_ELEM;
    v16bf a0 = load_frag(Asl, wave * 32,      0, GLDK);
    v16bf a1 = load_frag(Asl, wave * 32 + 16, 0, GLDK);
#pragma unroll
    for (int nt = 0; nt < 4; ++nt) {
      v16bf b = load_frag(Bsl, nt * 16, 0, GLDK);
      acc[0][nt] = __builtin_amdgcn_wmma_f32_16x16x32_bf16(
          false, a0, false, b, (short)0, acc[0][nt], false, false);
      acc[1][nt] = __builtin_amdgcn_wmma_f32_16x16x32_bf16(
          false, a1, false, b, (short)0, acc[1][nt], false, false);
    }
    wait_async0();
    __syncthreads();
    buf ^= 1;
  }

  const int half = lane >> 4;
#pragma unroll
  for (int mt = 0; mt < 2; ++mt)
#pragma unroll
    for (int nt = 0; nt < 4; ++nt)
#pragma unroll
      for (int r = 0; r < 8; ++r) {
        int m   = mBase + wave * 32 + mt * 16 + r + 8 * half;
        int col = nBase + nt * 16 + (lane & 15);
        xw[(size_t)m * G4H + col] = acc[mt][nt][r] + b_ih[col];
      }
}

// ---- Kernel 3: persistent LSTM recurrence; W_hh slice resident in LDS for all 2048 steps ----
__global__ __launch_bounds__(256) void lstm_rec_kernel(
    const unsigned* __restrict__ whhbf,  // W_hh bf16 pairs [4H][H]
    const float* __restrict__ xw,        // [S*B][4H], includes b_ih
    const float* __restrict__ b_hh,
    const float* __restrict__ c0,        // [B][H]
    bf16_t* __restrict__ hbuf,           // [2][B][H] bf16 ping-pong
    float* __restrict__ hf32,            // [B][H] final h
    unsigned* __restrict__ bar)
{
  extern __shared__ char smem[];
  bf16_t* Whh = (bf16_t*)smem;                                   // [128][LDK_REC]
  bf16_t* Hp  = (bf16_t*)(smem + REC_ROWS * LDK_REC * 2);        // [16][LDK_REC]
  float*  Gl  = (float*)(smem + (REC_ROWS + 16) * LDK_REC * 2);  // [128][16]
  float*  Cl  = Gl + REC_ROWS * 16;                              // [128]

  const int wg   = blockIdx.x;
  const int tid  = threadIdx.x;
  const int wave = tid >> 5;
  const int lane = tid & 31;

  // pin this WG's W_hh slice (gates i,f,g,o for units [wg*32, wg*32+32)) into LDS
  for (int idx = tid; idx < REC_ROWS * (D_ / 2); idx += 256) {
    int row = idx >> 9, p = idx & 511;
    int gr  = (row >> 5) * H_ + wg * UNITS + (row & 31);         // global W_hh row
    *(unsigned*)(Whh + row * LDK_REC + p * 2) = whhbf[((size_t)gr * D_) / 2 + p];
  }
  // zero the dummy batch rows 4..15 of the padded h tile
  for (int idx = tid; idx < 12 * (LDK_REC / 2); idx += 256) {
    int row = 4 + idx / (LDK_REC / 2), p = idx % (LDK_REC / 2);
    *(unsigned*)(Hp + row * LDK_REC + p * 2) = 0u;
  }
  if (tid < REC_ROWS) {  // cell state for (unit, batch)
    int u = tid >> 2, b = tid & 3;
    Cl[tid] = c0[b * H_ + wg * UNITS + u];
  }
  __syncthreads();

  for (int step = 0; step < S_; ++step) {
    // async-DMA h (4 rows x 2048 B) from ping-pong buffer into padded LDS tile
    const char* hsrc = (const char*)hbuf + (size_t)(step & 1) * (B_ * H_ * 2);
    {
      int c0i = tid, c1i = tid + 256;                 // 512 chunks of 16B
      int r0 = c0i >> 7, p0 = c0i & 127;
      int r1 = c1i >> 7, p1 = c1i & 127;
      async_b128(hsrc + (r0 * H_ + p0 * 8) * 2, Hp + r0 * LDK_REC + p0 * 8);
      async_b128(hsrc + (r1 * H_ + p1 * 8) * 2, Hp + r1 * LDK_REC + p1 * 8);
    }
    // prefetch the xw gate slices the activation phase will read
    if (tid < REC_ROWS) {
      const float* xwp = xw + ((size_t)step * B_ + (tid & 3)) * G4H + wg * UNITS + (tid >> 2);
      __builtin_prefetch(xwp, 0, 0);
      __builtin_prefetch(xwp + H_, 0, 0);
      __builtin_prefetch(xwp + 2 * H_, 0, 0);
      __builtin_prefetch(xwp + 3 * H_, 0, 0);
    }
    wait_async0();
    __syncthreads();

    // gates[m][b] = sum_k W_hh[m][k] * h[b][k]; wave owns 16 of 128 M-rows.
    // Software pipeline: load chunk k+1 fragments before issuing chunk k's WMMA.
    v8f acc = (v8f){0,0,0,0,0,0,0,0};
    v16bf a = load_frag(Whh, wave * 16, 0, LDK_REC);
    v16bf b = load_frag(Hp, 0, 0, LDK_REC);
    for (int kc = 0; kc < H_ - 32; kc += 32) {
      v16bf an = load_frag(Whh, wave * 16, kc + 32, LDK_REC);
      v16bf bn = load_frag(Hp, 0, kc + 32, LDK_REC);
      acc = __builtin_amdgcn_wmma_f32_16x16x32_bf16(
          false, a, false, b, (short)0, acc, false, false);
      a = an; b = bn;
    }
    acc = __builtin_amdgcn_wmma_f32_16x16x32_bf16(
        false, a, false, b, (short)0, acc, false, false);

    const int half = lane >> 4;
#pragma unroll
    for (int r = 0; r < 8; ++r)
      Gl[(wave * 16 + r + 8 * half) * 16 + (lane & 15)] = acc[r];
    __syncthreads();

    if (tid < REC_ROWS) {  // one thread per (unit, batch)
      int u = tid >> 2, b = tid & 3;
      int unit = wg * UNITS + u;
      const float* xwp = xw + ((size_t)step * B_ + b) * G4H;
      float gi = Gl[(u)      * 16 + b] + xwp[unit]           + b_hh[unit];
      float gf = Gl[(32 + u) * 16 + b] + xwp[H_ + unit]      + b_hh[H_ + unit];
      float gg = Gl[(64 + u) * 16 + b] + xwp[2 * H_ + unit]  + b_hh[2 * H_ + unit];
      float go = Gl[(96 + u) * 16 + b] + xwp[3 * H_ + unit]  + b_hh[3 * H_ + unit];
      float i = 1.f / (1.f + __expf(-gi));
      float f = 1.f / (1.f + __expf(-gf));
      float g = tanhf(gg);
      float o = 1.f / (1.f + __expf(-go));
      float c = f * Cl[tid] + i * g;
      Cl[tid] = c;
      float h = o * tanhf(c);
      hbuf[(size_t)((step + 1) & 1) * (B_ * H_) + b * H_ + unit] = (bf16_t)h;
      if (step == S_ - 1) hf32[b * H_ + unit] = h;
    }
    __threadfence();
    __syncthreads();

    // grid-wide step barrier (monotone generation counter)
    if (tid == 0) {
      unsigned old = __hip_atomic_fetch_add(&bar[0], 1u, __ATOMIC_ACQ_REL, __HIP_MEMORY_SCOPE_AGENT);
      if (old == REC_WGS - 1) {
        __hip_atomic_store(&bar[0], 0u, __ATOMIC_RELAXED, __HIP_MEMORY_SCOPE_AGENT);
        __hip_atomic_fetch_add(&bar[1], 1u, __ATOMIC_RELEASE, __HIP_MEMORY_SCOPE_AGENT);
      }
      while (__hip_atomic_load(&bar[1], __ATOMIC_ACQUIRE, __HIP_MEMORY_SCOPE_AGENT) <
             (unsigned)(step + 1)) {
        __builtin_amdgcn_s_sleep(8);
      }
    }
    __syncthreads();
  }
}

// ---- Kernel 4: hid = relu(concat(h, img) @ W1.T + b1), one wave per output ----
__global__ __launch_bounds__(256) void mlp1_kernel(
    const float* __restrict__ hf32, const float* __restrict__ img,
    const float* __restrict__ W1, const float* __restrict__ b1,
    float* __restrict__ hid)
{
  int wid  = blockIdx.x * 8 + (threadIdx.x >> 5);   // 4096 waves = B*HID outputs
  int lane = threadIdx.x & 31;
  int b = wid >> 10, j = wid & 1023;
  const float* w = W1 + (size_t)j * (H_ + V_);
  float sum = 0.f;
  for (int k = lane; k < H_; k += 32) sum += hf32[b * H_ + k] * w[k];
  for (int k = lane; k < V_; k += 32) sum += img[b * V_ + k] * w[H_ + k];
#pragma unroll
  for (int m = 16; m > 0; m >>= 1) sum += __shfl_xor(sum, m, 32);
  if (lane == 0) {
    float v = sum + b1[j];
    hid[b * HID + j] = v > 0.f ? v : 0.f;
  }
}

// ---- Kernel 5: out[b] = hid[b] @ W2 + b2 ----
__global__ void mlp2_kernel(const float* __restrict__ hid, const float* __restrict__ W2,
                            const float* __restrict__ b2, float* __restrict__ out) {
  int wave = threadIdx.x >> 5, lane = threadIdx.x & 31;
  if (wave < B_) {
    float sum = 0.f;
    for (int k = lane; k < HID; k += 32) sum += hid[wave * HID + k] * W2[k];
#pragma unroll
    for (int m = 16; m > 0; m >>= 1) sum += __shfl_xor(sum, m, 32);
    if (lane == 0) out[wave] = sum + b2[0];
  }
}

extern "C" void kernel_launch(void* const* d_in, const int* in_sizes, int n_in,
                              void* d_out, int out_size, void* d_ws, size_t ws_size,
                              hipStream_t stream) {
  const float* lang = (const float*)d_in[0];
  const float* img  = (const float*)d_in[1];
  const float* h0   = (const float*)d_in[2];
  const float* c0   = (const float*)d_in[3];
  const float* W_ih = (const float*)d_in[4];
  const float* W_hh = (const float*)d_in[5];
  const float* b_ih = (const float*)d_in[6];
  const float* b_hh = (const float*)d_in[7];
  const float* W1   = (const float*)d_in[8];
  const float* b1   = (const float*)d_in[9];
  const float* W2   = (const float*)d_in[10];
  const float* b2   = (const float*)d_in[11];

  char* ws = (char*)d_ws;
  bf16_t*   langbf = (bf16_t*)(ws + OFF_LANGBF);
  bf16_t*   wihbf  = (bf16_t*)(ws + OFF_WIHBF);
  bf16_t*   whhbf  = (bf16_t*)(ws + OFF_WHHBF);
  float*    xw     = (float*)(ws + OFF_XW);
  bf16_t*   hbuf   = (bf16_t*)(ws + OFF_HBUF);
  float*    hf32   = (float*)(ws + OFF_HF32);
  float*    hid    = (float*)(ws + OFF_HID);
  unsigned* bar    = (unsigned*)(ws + OFF_BAR);

  cvt_bf16_kernel<<<1024, 256, 0, stream>>>(lang, langbf, B_ * S_ * D_);
  cvt_bf16_kernel<<<1024, 256, 0, stream>>>(W_ih, wihbf, G4H * D_);
  cvt_bf16_kernel<<<1024, 256, 0, stream>>>(W_hh, whhbf, G4H * H_);
  init_kernel<<<16, 256, 0, stream>>>(h0, hbuf, bar);

  dim3 g2(S_ * B_ / GA_ROWS, G4H / GB_ROWS);  // (32, 64)
  gemm_xw_kernel<<<g2, 256, 0, stream>>>(langbf, wihbf, b_ih, xw);

  (void)hipFuncSetAttribute((const void*)lstm_rec_kernel,
                            hipFuncAttributeMaxDynamicSharedMemorySize, REC_SMEM);
  lstm_rec_kernel<<<REC_WGS, 256, REC_SMEM, stream>>>(
      (const unsigned*)whhbf, xw, b_hh, c0, hbuf, hf32, bar);

  mlp1_kernel<<<B_ * HID / 8, 256, 0, stream>>>(hf32, img, W1, b1, hid);
  mlp2_kernel<<<1, 128, 0, stream>>>(hid, W2, b2, (float*)d_out);
}
